// GraphTransformerModule_64037962384024
// MI455X (gfx1250) — compile-verified, matrix operand order checked by reading, not run on hardware
//
#include <hip/hip_runtime.h>

// ---------------------------------------------------------------------------
// GraphTransformer layer for MI455X (gfx1250), wave32 + WMMA bf16.
// Pipeline:
//   cvt weights->bf16 (once, L2-resident) -> zero(wV,z) -> qkv gemm ->
//   fused edge kernel (proj_e + score + exp + scatter-atomics + edge MLP) ->
//   h = wV/(z+eps) -> node MLP.
// All GEMMs use v_wmma_f32_16x16x32_bf16 with fp32 accumulation.
// Global accesses use explicit address_space(1) so they lower to
// global_load/global_store (not flat_*), decoupling LOADcnt from DScnt.
// ---------------------------------------------------------------------------

#define GAS __attribute__((address_space(1)))

typedef __attribute__((ext_vector_type(16))) __bf16 v16bf;
typedef __attribute__((ext_vector_type(8)))  __bf16 v8bf;
typedef __attribute__((ext_vector_type(8)))  float  v8f;
typedef __attribute__((ext_vector_type(4)))  float  v4f;   // plain vector (no HIP class)

__device__ __forceinline__ v8f zero8() {
  v8f z = {0.f, 0.f, 0.f, 0.f, 0.f, 0.f, 0.f, 0.f};
  return z;
}

__device__ __forceinline__ v8f wmma_bf16(v16bf a, v16bf b, v8f c) {
  // 8 args: (neg_a, A, neg_b, B, c_mod, C, reuse_a, reuse_b)
  return __builtin_amdgcn_wmma_f32_16x16x32_bf16(false, a, false, b,
                                                 (short)0, c, false, false);
}

// ---- global-AS load/store helpers (force global_*, not flat_*) ----
__device__ __forceinline__ v4f ld_g4(const float* p) {
  return *(const GAS v4f*)p;
}
__device__ __forceinline__ float ld_gf(const float* p) {
  return *(const GAS float*)p;
}
__device__ __forceinline__ int ld_gi(const int* p) {
  return *(const GAS int*)p;
}
__device__ __forceinline__ void st_gf(float* p, float v) {
  *(GAS float*)p = v;
}
__device__ __forceinline__ void st_g4(float* p, v4f v) {
  *(GAS v4f*)p = v;
}
__device__ __forceinline__ void st_gh(__bf16* p, __bf16 v) {
  *(GAS __bf16*)p = v;
}

// 16x32 bf16 fragment from GLOBAL bf16 (A layout; row-major [N,K] weights
// are exactly the B=A^T layout). ISA 7.12.2: half = lane>>4, m = lane&15,
// k(i) = half*8 + (i<8 ? i : i+8)  -> two contiguous 16B halves.
__device__ __forceinline__ v16bf frag_bf16_g(const __bf16* base, int ld,
                                             int lane) {
  const int half = (lane >> 4) & 1;
  const int m    = lane & 15;
  const __bf16* p = base + m * ld + half * 8;
  union { v16bf v; v8bf h[2]; } u;
  u.h[0] = *(const GAS v8bf*)p;
  u.h[1] = *(const GAS v8bf*)(p + 16);
  return u.v;
}

// Same fragment gather from LDS (ds_load_b128 x2).
__device__ __forceinline__ v16bf frag_bf16_lds(const __bf16* base, int ld,
                                               int lane) {
  const int half = (lane >> 4) & 1;
  const int m    = lane & 15;
  const __bf16* p = base + m * ld + half * 8;
  union { v16bf v; v8bf h[2]; } u;
  u.h[0] = *(const v8bf*)p;
  u.h[1] = *(const v8bf*)(p + 16);
  return u.v;
}

__device__ __forceinline__ float silu_f(float x) {
  return x / (1.f + __expf(-x));
}
__device__ __forceinline__ float clip5(float x) {
  return fminf(5.f, fmaxf(-5.f, x));
}

// Stage a 64x128 fp32 tile (row stride 128) into LDS as bf16, zero-padding
// out-of-range rows. 128 threads, float4 global loads.
__device__ __forceinline__ void stage_tile_64x128(const float* __restrict__ G,
                                                  int m0, int M,
                                                  __bf16* __restrict__ dstLds,
                                                  int tid) {
#pragma unroll
  for (int it = 0; it < 16; ++it) {
    int idx = tid + it * 128;       // float4 index, 2048 total
    int r   = idx >> 5;             // 32 float4 per row
    int c4  = (idx & 31) * 4;
    int gm  = m0 + r;
    v4f v = {0.f, 0.f, 0.f, 0.f};
    if (gm < M) v = ld_g4(G + (size_t)gm * 128 + c4);
    __bf16* p = dstLds + r * 128 + c4;
    p[0] = (__bf16)v.x; p[1] = (__bf16)v.y;
    p[2] = (__bf16)v.z; p[3] = (__bf16)v.w;
  }
}

// ---------------------------------------------------------------------------
__global__ void zero_kernel(float* __restrict__ p, int n) {
  int i = blockIdx.x * blockDim.x + threadIdx.x;
  if (i < n) st_gf(p + i, 0.f);
}

__global__ void cvt_bf16_kernel(const float* __restrict__ s,
                                __bf16* __restrict__ d, int n) {
  int i = blockIdx.x * blockDim.x + threadIdx.x;
  if (i < n) st_gh(d + i, (__bf16)ld_gf(s + i));
}

// Q,K,V = node_feats @ {Wq,Wk,Wv}^T  (M x 128, K=128), weights bf16.
__global__ void __launch_bounds__(128)
qkv_kernel(const float* __restrict__ X,
           const __bf16* __restrict__ WqH, const __bf16* __restrict__ WkH,
           const __bf16* __restrict__ WvH,
           float* __restrict__ Qo, float* __restrict__ Ko,
           float* __restrict__ Vo, int M) {
  __shared__ __align__(16) __bf16 ldsX[64 * 128];
  const int tid = threadIdx.x, lane = tid & 31, wave = tid >> 5;
  const int nl = lane & 15, half = lane >> 4;
  const int m0 = blockIdx.x * 64;

  stage_tile_64x128(X, m0, M, ldsX, tid);
  __syncthreads();

  const __bf16* Ws[3] = {WqH, WkH, WvH};
  float*        Os[3] = {Qo, Ko, Vo};

#pragma unroll
  for (int p = 0; p < 3; ++p) {
    v8f acc[8];
#pragma unroll
    for (int j = 0; j < 8; ++j) acc[j] = zero8();
#pragma unroll
    for (int k0 = 0; k0 < 128; k0 += 32) {
      v16bf a = frag_bf16_lds(ldsX + wave * 16 * 128 + k0, 128, lane);
#pragma unroll
      for (int j = 0; j < 8; ++j) {
        v16bf b = frag_bf16_g(Ws[p] + j * 16 * 128 + k0, 128, lane);
        acc[j] = wmma_bf16(a, b, acc[j]);
      }
    }
#pragma unroll
    for (int j = 0; j < 8; ++j) {
#pragma unroll
      for (int r = 0; r < 8; ++r) {
        int gm = m0 + wave * 16 + r + 8 * half;
        if (gm < M) st_gf(Os[p] + (size_t)gm * 128 + j * 16 + nl, acc[j][r]);
      }
    }
  }
}

// Fused edge kernel: proj_e GEMM -> score -> exp-weights -> scatter atomics
// -> edge MLP (128 -> 256 silu -> 128), 64 edges per block, all in LDS.
__global__ void __launch_bounds__(128)
edge_kernel(const float* __restrict__ EF, const int* __restrict__ src,
            const int* __restrict__ dst, const __bf16* __restrict__ WeH,
            const float* __restrict__ Qb, const float* __restrict__ Kb,
            const float* __restrict__ Vb, const __bf16* __restrict__ We1H,
            const __bf16* __restrict__ We2H, float* __restrict__ wV,
            float* __restrict__ zb, float* __restrict__ Eout, int E_) {
  // overlay[0..32KB): ldsX (stage 0) then hid (MLP); scoreH at +32KB.
  __shared__ __align__(16) unsigned char smem[50688];
  __bf16* overlay = (__bf16*)smem;                 // 64*256 bf16 = 32768 B
  __bf16* scoreH  = (__bf16*)(smem + 32768);       // 64*128 bf16 = 16384 B
  float*  sBuf    = (float*)(smem + 49152);        // 64*4 f32    = 1024 B
  int*    sIdx    = (int*)(smem + 50176);          // 64 ints
  int*    dIdx    = (int*)(smem + 50432);          // 64 ints

  const int tid = threadIdx.x, lane = tid & 31, wave = tid >> 5;
  const int nl = lane & 15, half = lane >> 4;
  const int e0 = blockIdx.x * 64;

  if (tid < 64) {
    int e = e0 + tid;
    int s = (e < E_) ? ld_gi(src + e) : 0;
    int d = (e < E_) ? ld_gi(dst + e) : 0;
    sIdx[tid] = s;
    dIdx[tid] = d;
    // warm the gathered rows (global_prefetch_b8)
    __builtin_prefetch(Kb + (size_t)s * 128, 0, 1);
    __builtin_prefetch(Qb + (size_t)d * 128, 0, 1);
    __builtin_prefetch(Vb + (size_t)s * 128, 0, 1);
  }

  // ---- stage 0: proj = edge_feats @ We^T (64 x 128) ----
  __bf16* ldsX = overlay;
  stage_tile_64x128(EF, e0, E_, ldsX, tid);
  __syncthreads();

  {
    v8f acc[8];
#pragma unroll
    for (int j = 0; j < 8; ++j) acc[j] = zero8();
#pragma unroll
    for (int k0 = 0; k0 < 128; k0 += 32) {
      v16bf a = frag_bf16_lds(ldsX + wave * 16 * 128 + k0, 128, lane);
#pragma unroll
      for (int j = 0; j < 8; ++j) {
        v16bf b = frag_bf16_g(WeH + j * 16 * 128 + k0, 128, lane);
        acc[j] = wmma_bf16(a, b, acc[j]);
      }
    }
#pragma unroll
    for (int j = 0; j < 8; ++j) {
#pragma unroll
      for (int r = 0; r < 8; ++r) {
        int row = wave * 16 + r + 8 * half;
        scoreH[row * 128 + j * 16 + nl] = (__bf16)acc[j][r];
      }
    }
  }
  __syncthreads();

  // ---- stage 1: score = clip(K[src]*Q[dst]/sqrt(dh)) * proj ----
  const float inv_scale = 0.17677669529663688f;  // 1/sqrt(32)
#pragma unroll 4
  for (int i = 0; i < 16; ++i) {
    int idx4 = tid + i * 128;       // 2048 float4 groups
    int er   = idx4 >> 5;           // 32 groups per row
    int c4   = (idx4 & 31) * 4;
    int e    = e0 + er;
    __bf16* p = scoreH + er * 128 + c4;
    float v0 = 0.f, v1 = 0.f, v2 = 0.f, v3 = 0.f;
    if (e < E_) {
      v4f kr = ld_g4(Kb + (size_t)sIdx[er] * 128 + c4);
      v4f qr = ld_g4(Qb + (size_t)dIdx[er] * 128 + c4);
      v0 = clip5(kr.x * qr.x * inv_scale) * (float)p[0];
      v1 = clip5(kr.y * qr.y * inv_scale) * (float)p[1];
      v2 = clip5(kr.z * qr.z * inv_scale) * (float)p[2];
      v3 = clip5(kr.w * qr.w * inv_scale) * (float)p[3];
    }
    p[0] = (__bf16)v0; p[1] = (__bf16)v1;
    p[2] = (__bf16)v2; p[3] = (__bf16)v3;
  }
  __syncthreads();

  // ---- stage 2: s = exp(clip(sum_dh score)); z[dst] += s ----
#pragma unroll
  for (int i = 0; i < 2; ++i) {
    int idx = tid + i * 128;        // 256 (edge, head) pairs
    int er = idx >> 2, hh = idx & 3;
    float sum = 0.f;
#pragma unroll
    for (int d = 0; d < 32; ++d) sum += (float)scoreH[er * 128 + hh * 32 + d];
    float s = __expf(clip5(sum));
    sBuf[er * 4 + hh] = s;
    int e = e0 + er;
    if (e < E_) atomicAdd(&zb[(size_t)dIdx[er] * 4 + hh], s);
  }
  __syncthreads();

  // ---- stage 3: wV[dst] += V[src] * s ----
  for (int i = 0; i < 64; ++i) {
    int idx = tid + i * 128;
    int er = idx >> 7, c = idx & 127;
    int e = e0 + er;
    if (e < E_) {
      float s = sBuf[er * 4 + (c >> 5)];
      atomicAdd(&wV[(size_t)dIdx[er] * 128 + c],
                ld_gf(Vb + (size_t)sIdx[er] * 128 + c) * s);
    }
  }

  // ---- edge MLP stage 1: hid = silu(score @ We1^T), 64 x 256 ----
  __bf16* hid = overlay;   // overlays dead ldsX (barriers passed since)
  {
    v8f h1[16];
#pragma unroll
    for (int j = 0; j < 16; ++j) h1[j] = zero8();
#pragma unroll
    for (int k0 = 0; k0 < 128; k0 += 32) {
      v16bf a = frag_bf16_lds(scoreH + wave * 16 * 128 + k0, 128, lane);
#pragma unroll
      for (int j = 0; j < 16; ++j) {
        v16bf b = frag_bf16_g(We1H + j * 16 * 128 + k0, 128, lane);
        h1[j] = wmma_bf16(a, b, h1[j]);
      }
    }
#pragma unroll
    for (int j = 0; j < 16; ++j) {
#pragma unroll
      for (int r = 0; r < 8; ++r) {
        int row = wave * 16 + r + 8 * half;
        hid[row * 256 + j * 16 + nl] = (__bf16)silu_f(h1[j][r]);
      }
    }
  }
  __syncthreads();

  // ---- edge MLP stage 2: out = hid @ We2^T, 64 x 128, K=256 ----
  {
    v8f o[8];
#pragma unroll
    for (int j = 0; j < 8; ++j) o[j] = zero8();
#pragma unroll
    for (int k0 = 0; k0 < 256; k0 += 32) {
      v16bf a = frag_bf16_lds(hid + wave * 16 * 256 + k0, 256, lane);
#pragma unroll
      for (int j = 0; j < 8; ++j) {
        v16bf b = frag_bf16_g(We2H + j * 16 * 256 + k0, 256, lane);
        o[j] = wmma_bf16(a, b, o[j]);
      }
    }
#pragma unroll
    for (int j = 0; j < 8; ++j) {
#pragma unroll
      for (int r = 0; r < 8; ++r) {
        int e = e0 + wave * 16 + r + 8 * half;
        if (e < E_) st_gf(Eout + (size_t)e * 128 + j * 16 + nl, o[j][r]);
      }
    }
  }
}

// h = wV / (z + 1e-6), float4 per thread
__global__ void h_kernel(const float* __restrict__ wV,
                         const float* __restrict__ zb,
                         float* __restrict__ h, int total4) {
  int i = blockIdx.x * blockDim.x + threadIdx.x;
  if (i < total4) {
    int n  = i >> 5;                // 32 float4 per node row
    int c4 = (i & 31) * 4;
    float zz = ld_gf(zb + n * 4 + (c4 >> 5)) + 1e-6f;
    v4f w = ld_g4(wV + (size_t)i * 4);
    v4f r = {w.x / zz, w.y / zz, w.z / zz, w.w / zz};
    st_g4(h + (size_t)i * 4, r);
  }
}

// Out = silu(X @ W1^T) @ W2^T    X:[M,128], W1:[256,128], W2:[128,256] (bf16)
__global__ void __launch_bounds__(128)
mlp_kernel(const float* __restrict__ X, const __bf16* __restrict__ W1H,
           const __bf16* __restrict__ W2H, float* __restrict__ Out, int M) {
  __shared__ __align__(16) __bf16 ldsX[64 * 128];
  __shared__ __align__(16) __bf16 hid[64 * 256];
  const int tid = threadIdx.x, lane = tid & 31, wave = tid >> 5;
  const int nl = lane & 15, half = lane >> 4;
  const int m0 = blockIdx.x * 64;

  stage_tile_64x128(X, m0, M, ldsX, tid);
  __syncthreads();

  {
    v8f h1[16];
#pragma unroll
    for (int j = 0; j < 16; ++j) h1[j] = zero8();
#pragma unroll
    for (int k0 = 0; k0 < 128; k0 += 32) {
      v16bf a = frag_bf16_lds(ldsX + wave * 16 * 128 + k0, 128, lane);
#pragma unroll
      for (int j = 0; j < 16; ++j) {
        v16bf b = frag_bf16_g(W1H + j * 16 * 128 + k0, 128, lane);
        h1[j] = wmma_bf16(a, b, h1[j]);
      }
    }
#pragma unroll
    for (int j = 0; j < 16; ++j) {
#pragma unroll
      for (int r = 0; r < 8; ++r) {
        int row = wave * 16 + r + 8 * half;
        hid[row * 256 + j * 16 + nl] = (__bf16)silu_f(h1[j][r]);
      }
    }
  }
  __syncthreads();

  {
    v8f o[8];
#pragma unroll
    for (int j = 0; j < 8; ++j) o[j] = zero8();
#pragma unroll
    for (int k0 = 0; k0 < 256; k0 += 32) {
      v16bf a = frag_bf16_lds(hid + wave * 16 * 256 + k0, 256, lane);
#pragma unroll
      for (int j = 0; j < 8; ++j) {
        v16bf b = frag_bf16_g(W2H + j * 16 * 256 + k0, 256, lane);
        o[j] = wmma_bf16(a, b, o[j]);
      }
    }
#pragma unroll
    for (int j = 0; j < 8; ++j) {
#pragma unroll
      for (int r = 0; r < 8; ++r) {
        int gm = m0 + wave * 16 + r + 8 * half;
        if (gm < M) st_gf(Out + (size_t)gm * 128 + j * 16 + nl, o[j][r]);
      }
    }
  }
}

// ---------------------------------------------------------------------------
extern "C" void kernel_launch(void* const* d_in, const int* in_sizes, int n_in,
                              void* d_out, int out_size, void* d_ws,
                              size_t ws_size, hipStream_t stream) {
  const float* node_feats = (const float*)d_in[0];
  const float* edge_feats = (const float*)d_in[1];
  const int*   src        = (const int*)d_in[2];
  const int*   dst        = (const int*)d_in[3];
  const float* Wq         = (const float*)d_in[4];
  const float* Wk         = (const float*)d_in[5];
  const float* Wv         = (const float*)d_in[6];
  const float* We         = (const float*)d_in[7];
  const float* Wn1        = (const float*)d_in[8];
  const float* Wn2        = (const float*)d_in[9];
  const float* We1        = (const float*)d_in[10];
  const float* We2        = (const float*)d_in[11];

  const int N = in_sizes[0] / 128;
  const int E = in_sizes[1] / 128;

  float* ws = (float*)d_ws;
  float* Qb = ws;
  float* Kb = Qb + (size_t)N * 128;
  float* Vb = Kb + (size_t)N * 128;
  float* wV = Vb + (size_t)N * 128;
  float* zb = wV + (size_t)N * 128;   // N*4
  float* hb = zb + (size_t)N * 4;

  // bf16 weight cache (L2-resident, ~384 KB)
  __bf16* wH   = (__bf16*)(hb + (size_t)N * 128);
  __bf16* WqH  = wH;
  __bf16* WkH  = WqH + 16384;
  __bf16* WvH  = WkH + 16384;
  __bf16* WeH  = WvH + 16384;
  __bf16* Wn1H = WeH + 16384;
  __bf16* Wn2H = Wn1H + 32768;
  __bf16* We1H = Wn2H + 32768;
  __bf16* We2H = We1H + 32768;

  float* node_out = (float*)d_out;
  float* edge_out = node_out + (size_t)N * 128;

  // convert weights fp32 -> bf16 (cheap; runs every launch, deterministic)
  cvt_bf16_kernel<<<64, 256, 0, stream>>>(Wq, WqH, 16384);
  cvt_bf16_kernel<<<64, 256, 0, stream>>>(Wk, WkH, 16384);
  cvt_bf16_kernel<<<64, 256, 0, stream>>>(Wv, WvH, 16384);
  cvt_bf16_kernel<<<64, 256, 0, stream>>>(We, WeH, 16384);
  cvt_bf16_kernel<<<128, 256, 0, stream>>>(Wn1, Wn1H, 32768);
  cvt_bf16_kernel<<<128, 256, 0, stream>>>(Wn2, Wn2H, 32768);
  cvt_bf16_kernel<<<128, 256, 0, stream>>>(We1, We1H, 32768);
  cvt_bf16_kernel<<<128, 256, 0, stream>>>(We2, We2H, 32768);

  // zero wV and z (contiguous: N*128 + N*4 floats)
  int zlen = N * 132;
  zero_kernel<<<(zlen + 255) / 256, 256, 0, stream>>>(wV, zlen);

  qkv_kernel<<<(N + 63) / 64, 128, 0, stream>>>(node_feats, WqH, WkH, WvH,
                                                Qb, Kb, Vb, N);

  edge_kernel<<<(E + 63) / 64, 128, 0, stream>>>(edge_feats, src, dst, WeH,
                                                 Qb, Kb, Vb, We1H, We2H,
                                                 wV, zb, edge_out, E);

  h_kernel<<<((N * 32) + 255) / 256, 256, 0, stream>>>(wV, zb, hb, N * 32);

  mlp_kernel<<<(N + 63) / 64, 128, 0, stream>>>(hb, Wn1H, Wn2H, node_out, N);
}
